// CharRNN_16801912062005
// MI455X (gfx1250) — matrix operand order repaired and checked by reading, not compiled, not run on hardware
//
#include <hip/hip_runtime.h>
#include <hip/hip_fp16.h>

typedef __attribute__((ext_vector_type(16))) _Float16 v16h;
typedef __attribute__((ext_vector_type(8)))  _Float16 v8h;
typedef __attribute__((ext_vector_type(8)))  float    v8f;

#define B_   256
#define L_   1024
#define E_   128
#define H_   256
#define V_   64
#define ROWS 16   // batch rows handled per workgroup

// ---------------------------------------------------------------------------
// Prep: P[v,n] = emb[v,:]·W_e[n,:] + b_e[n] + b_h[n]  (the fused input proj),
// plus fp16 copies of W_h and fc_w for the WMMA pipeline.
// grid = 64 (P) + 256 (W_h) + 64 (fc_w) = 384 blocks of 256 threads.
// ---------------------------------------------------------------------------
__global__ __launch_bounds__(256) void prep_tables(
    const float* __restrict__ emb,   // [V,E]
    const float* __restrict__ W_e,   // [H,E]
    const float* __restrict__ b_e,   // [H]
    const float* __restrict__ b_h,   // [H]
    const float* __restrict__ W_h,   // [H,H]
    const float* __restrict__ fc_w,  // [V,H]
    float*       __restrict__ P,     // [V,H]
    _Float16*    __restrict__ Wh16,  // [H,H]
    _Float16*    __restrict__ fcw16) // [V,H]
{
    int blk = blockIdx.x;
    if (blk < V_) {
        int v = blk, n = threadIdx.x;
        float s = b_e[n] + b_h[n];
        const float* ev = emb + v * E_;
        const float* wn = W_e + n * E_;
        #pragma unroll 4
        for (int e = 0; e < E_; ++e) s += ev[e] * wn[e];
        P[v * H_ + n] = s;
    } else if (blk < V_ + 256) {
        int i = (blk - V_) * 256 + threadIdx.x;        // 65536 elements
        Wh16[i] = (_Float16)W_h[i];
    } else {
        int i = (blk - V_ - 256) * 256 + threadIdx.x;  // 16384 elements
        fcw16[i] = (_Float16)fc_w[i];
    }
}

__device__ __forceinline__ v16h pack16(v8h lo, v8h hi) {
    return __builtin_shufflevector(lo, hi, 0,1,2,3,4,5,6,7,8,9,10,11,12,13,14,15);
}

// ---------------------------------------------------------------------------
// Persistent recurrence kernel: 16 WGs x 256 threads (8 waves).
// Each WG owns 16 batch rows; all weights live in LDS (~262 KB / 320 KB).
// Per step: 8 waves x 2 N-tiles x 8 K-steps of v_wmma_f32_16x16x32_f16 for
// h_t, then 4 waves compute the fused 16x64 logits tile.
// ---------------------------------------------------------------------------
__global__ __launch_bounds__(256) void rnn_scan_kernel(
    const int*      __restrict__ x,       // [B,L] token ids (< 64)
    const float*    __restrict__ hidden0, // [B,H]
    const float*    __restrict__ fc_bg,   // [V]
    const float*    __restrict__ Pg,      // [V,H]
    const _Float16* __restrict__ Whg,     // [H,H]
    const _Float16* __restrict__ fcwg,    // [V,H]
    float*          __restrict__ logits,  // [B,L,V]
    float*          __restrict__ hfinal)  // [B,H]
{
    __shared__ _Float16      Wh [H_][H_];        // 128 KB  B-matrix: Wh[n][k]
    __shared__ _Float16      FCW[V_][H_];        //  32 KB  B-matrix: FCW[v][k]
    __shared__ float         P  [V_][H_];        //  64 KB  fused input proj (+b_h)
    __shared__ unsigned char IDS[ROWS][L_];      //  16 KB  token ids for our rows
    __shared__ _Float16      Hbuf[2][ROWS][H_];  //  16 KB  double-buffered h (f16)
    __shared__ float         FCB[V_];            // 256 B

    const int tid  = threadIdx.x;
    const int wave = tid >> 5;
    const int lane = tid & 31;
    const int nsub = lane & 15;   // N (or M for A-loads) within a 16-tile
    const int hi   = lane >> 4;   // lane half selects K/M sub-range
    const int b0   = blockIdx.x * ROWS;

    // ---- one-time LDS fill --------------------------------------------------
    for (int i = tid; i < H_ * H_; i += 256) ((_Float16*)Wh)[i]  = Whg[i];
    for (int i = tid; i < V_ * H_; i += 256) ((_Float16*)FCW)[i] = fcwg[i];
    for (int i = tid; i < V_ * H_; i += 256) ((float*)P)[i]      = Pg[i];
    for (int i = tid; i < V_;      i += 256) FCB[i]              = fc_bg[i];
    for (int i = tid; i < ROWS * L_; i += 256) {
        int m = i >> 10, t = i & (L_ - 1);
        IDS[m][t] = (unsigned char)x[(b0 + m) * L_ + t];
    }
    for (int i = tid; i < ROWS * H_; i += 256) {
        int m = i >> 8, n = i & (H_ - 1);
        Hbuf[0][m][n] = (_Float16)hidden0[(b0 + m) * H_ + n];
    }
    __syncthreads();

    const int n0 = wave * 32;   // this wave's 32-wide slice of H (two N-tiles)

    for (int t = 0; t < L_; ++t) {
        const int p  = t & 1;
        const int pn = p ^ 1;

        // -- init acc with fused input projection (LUT gather from LDS) ------
        v8f acc0, acc1;
        #pragma unroll
        for (int r = 0; r < 8; ++r) {
            int m   = r + hi * 8;                 // C/D layout: M = r (+8 hi half)
            int vid = IDS[m][t];
            acc0[r] = P[vid][n0 + nsub];
            acc1[r] = P[vid][n0 + 16 + nsub];
        }

        // -- h_{t-1} @ W_h^T via WMMA ----------------------------------------
        #pragma unroll
        for (int k0 = 0; k0 < H_; k0 += 32) {
            // A tile (16x32 f16): lane = M, VGPR0-3 = K ka..ka+7, VGPR4-7 = K+16
            int ka = k0 + hi * 8;
            v8h alo = *(const v8h*)&Hbuf[p][nsub][ka];
            v8h ahi = *(const v8h*)&Hbuf[p][nsub][ka + 16];
            v16h a  = pack16(alo, ahi);
            // B tiles (32x16 f16): lane = N, VGPR0-3 = K kb..kb+7, VGPR4-7 = K+16
            int kb = k0 + hi * 8;
            v16h bv0 = pack16(*(const v8h*)&Wh[n0 + nsub][kb],
                              *(const v8h*)&Wh[n0 + nsub][kb + 16]);
            v16h bv1 = pack16(*(const v8h*)&Wh[n0 + 16 + nsub][kb],
                              *(const v8h*)&Wh[n0 + 16 + nsub][kb + 16]);
            acc0 = __builtin_amdgcn_wmma_f32_16x16x32_f16(false, a, false, bv0,
                                                          (short)0, acc0, false, false);
            acc1 = __builtin_amdgcn_wmma_f32_16x16x32_f16(false, a, false, bv1,
                                                          (short)0, acc1, false, false);
        }

        // -- tanh and publish h_t into the other buffer ----------------------
        #pragma unroll
        for (int r = 0; r < 8; ++r) {
            int m = r + hi * 8;
            Hbuf[pn][m][n0 + nsub]      = (_Float16)tanhf(acc0[r]);
            Hbuf[pn][m][n0 + 16 + nsub] = (_Float16)tanhf(acc1[r]);
        }
        __syncthreads();   // single barrier per step (double-buffered h)

        // -- fused logits GEMM: 16(batch) x 64(vocab), waves 0..3 ------------
        if (wave < 4) {
            int v0 = wave * 16;
            v8f lac;
            float fb = FCB[v0 + nsub];
            #pragma unroll
            for (int r = 0; r < 8; ++r) lac[r] = fb;
            #pragma unroll
            for (int k0 = 0; k0 < H_; k0 += 32) {
                int ka = k0 + hi * 8;
                v16h a = pack16(*(const v8h*)&Hbuf[pn][nsub][ka],
                                *(const v8h*)&Hbuf[pn][nsub][ka + 16]);
                int kb = k0 + hi * 8;
                v16h bv = pack16(*(const v8h*)&FCW[v0 + nsub][kb],
                                 *(const v8h*)&FCW[v0 + nsub][kb + 16]);
                lac = __builtin_amdgcn_wmma_f32_16x16x32_f16(false, a, false, bv,
                                                             (short)0, lac, false, false);
            }
            #pragma unroll
            for (int r = 0; r < 8; ++r) {
                int m = r + hi * 8;
                logits[((size_t)(b0 + m) * L_ + t) * V_ + v0 + nsub] = lac[r];
            }
        }
    }

    // final hidden lives in Hbuf[0] after t = L-1 (p=1 wrote pn=0)
    for (int i = tid; i < ROWS * H_; i += 256) {
        int m = i >> 8, n = i & (H_ - 1);
        hfinal[(b0 + m) * H_ + n] = (float)Hbuf[0][m][n];
    }
}

// ---------------------------------------------------------------------------
extern "C" void kernel_launch(void* const* d_in, const int* in_sizes, int n_in,
                              void* d_out, int out_size, void* d_ws, size_t ws_size,
                              hipStream_t stream) {
    const int*   x      = (const int*)  d_in[0];
    const float* hidden = (const float*)d_in[1];
    const float* emb    = (const float*)d_in[2];
    const float* W_e    = (const float*)d_in[3];
    const float* b_e    = (const float*)d_in[4];
    const float* W_h    = (const float*)d_in[5];
    const float* b_h    = (const float*)d_in[6];
    const float* fc_w   = (const float*)d_in[7];
    const float* fc_b   = (const float*)d_in[8];

    float*    P     = (float*)d_ws;                              // 64 KB
    _Float16* Wh16  = (_Float16*)((char*)d_ws + 65536);          // 128 KB
    _Float16* fcw16 = (_Float16*)((char*)d_ws + 65536 + 131072); // 32 KB

    float* logits = (float*)d_out;                               // [B,L,V]
    float* hfinal = logits + (size_t)B_ * L_ * V_;               // [B,H]

    prep_tables<<<384, 256, 0, stream>>>(emb, W_e, b_e, b_h, W_h, fc_w,
                                         P, Wh16, fcw16);
    rnn_scan_kernel<<<B_ / ROWS, 256, 0, stream>>>(x, hidden, fc_b,
                                                   P, Wh16, fcw16,
                                                   logits, hfinal);
}
